// ConvNeXt_Block_17583596110313
// MI455X (gfx1250) — matrix operand order repaired and checked by reading, not compiled, not run on hardware
//
#include <hip/hip_runtime.h>
#include <hip/hip_bf16.h>
#include <stdint.h>

typedef __attribute__((ext_vector_type(16))) __bf16 v16bf;
typedef __attribute__((ext_vector_type(8)))  __bf16 v8bf;
typedef __attribute__((ext_vector_type(8)))  float  v8f;

#define BATCH   16
#define HEIGHT  56
#define WIDTH   56
#define C_IN    384
#define C_HID   1536
#define NTOK    (BATCH * HEIGHT * WIDTH)   // 50176
#define MTILE   64
#define NH      64                          // hidden chunk
#define LDY     (C_IN + 8)                  // 392 (x2B = 784B, 16B-divisible)
#define LDH     (NH + 8)                    // 72  (x2B = 144B, 16B-divisible)

// ---------------- weight transpose + bf16 convert ----------------
__global__ void transpose_w1_kernel(const float* __restrict__ w1, __bf16* __restrict__ w1t) {
    int i = blockIdx.x * 256 + threadIdx.x;            // over 1536*384, w1t[d][c]
    if (i >= C_IN * C_HID) return;
    int d = i / C_IN;
    int c = i - d * C_IN;
    w1t[i] = (__bf16)w1[(long)c * C_HID + d];
}

__global__ void transpose_w2_kernel(const float* __restrict__ w2, __bf16* __restrict__ w2t) {
    int i = blockIdx.x * 256 + threadIdx.x;            // over 384*1536, w2t[c][d]
    if (i >= C_IN * C_HID) return;
    int c = i / C_HID;
    int d = i - c * C_HID;
    w2t[i] = (__bf16)w2[(long)d * C_IN + c];
}

// ---------------- depthwise 7x7 conv + bias + LayerNorm -> bf16 tokens ----------------
__global__ __launch_bounds__(384) void dwconv_ln_kernel(
    const float* __restrict__ x, const float* __restrict__ kern,
    const float* __restrict__ kbias, const float* __restrict__ lns,
    const float* __restrict__ lnb, __bf16* __restrict__ y_out)
{
    __shared__ float s_sum[C_IN];
    __shared__ float s_sq[C_IN];

    const int c = threadIdx.x;
    const int p = blockIdx.x;
    const int b   = p / (HEIGHT * WIDTH);
    const int rem = p - b * (HEIGHT * WIDTH);
    const int h   = rem / WIDTH;
    const int w   = rem - h * WIDTH;

    float acc = kbias[c];
    #pragma unroll
    for (int kh = 0; kh < 7; ++kh) {
        int ih = h + kh - 3;
        if (ih < 0 || ih >= HEIGHT) continue;
        #pragma unroll
        for (int kw = 0; kw < 7; ++kw) {
            int iw = w + kw - 3;
            if (iw < 0 || iw >= WIDTH) continue;
            float xv = x[(((long)b * HEIGHT + ih) * WIDTH + iw) * C_IN + c];
            float kv = kern[(kh * 7 + kw) * C_IN + c];
            acc = fmaf(xv, kv, acc);
        }
    }

    s_sum[c] = acc;
    s_sq[c]  = acc * acc;
    __syncthreads();
    if (c < 128) {           // 384 -> 128
        s_sum[c] += s_sum[c + 128] + s_sum[c + 256];
        s_sq[c]  += s_sq[c + 128]  + s_sq[c + 256];
    }
    __syncthreads();
    for (int off = 64; off > 0; off >>= 1) {
        if (c < off) {
            s_sum[c] += s_sum[c + off];
            s_sq[c]  += s_sq[c + off];
        }
        __syncthreads();
    }
    const float mean = s_sum[0] * (1.0f / C_IN);
    const float var  = s_sq[0] * (1.0f / C_IN) - mean * mean;
    const float inv  = rsqrtf(var + 1e-6f);
    const float yn   = (acc - mean) * inv * lns[c] + lnb[c];
    y_out[(long)p * C_IN + c] = (__bf16)yn;
}

// ---------------- fused MLP: gelu(y@w1+b1)@w2+b2, *gamma, +x ----------------
__device__ __forceinline__ float gelu_tanh(float v) {
    float u = v * (0.7978845608028654f + 0.7978845608028654f * 0.044715f * v * v);
    return 0.5f * v * (1.0f + tanhf(u));
}

__global__ __launch_bounds__(256) void mlp_wmma_kernel(
    const __bf16* __restrict__ y, const __bf16* __restrict__ w1t,
    const float* __restrict__ b1, const __bf16* __restrict__ w2t,
    const float* __restrict__ b2, const float* __restrict__ gamma,
    const float* __restrict__ x, float* __restrict__ out)
{
    __shared__ __bf16 y_tile[MTILE * LDY];   // 64 x 384 (+pad)
    __shared__ __bf16 h_tile[MTILE * LDH];   // 64 x 64  (+pad)

    const int tid  = threadIdx.x;
    const int lane = tid & 31;
    const int wv   = tid >> 5;        // 0..7
    const int mw   = wv & 3;          // M wave block (4 x 16 rows)
    const int nw   = wv >> 2;         // N wave half  (2)
    const long tok0 = (long)blockIdx.x * MTILE;

    // stage y tile into LDS with 8-byte vectors
    {
        const uint64_t* src64 = (const uint64_t*)(y + tok0 * C_IN);
        for (int i = tid; i < MTILE * (C_IN / 4); i += 256) {
            int r   = i / (C_IN / 4);
            int kk4 = i - r * (C_IN / 4);
            *(uint64_t*)(y_tile + r * LDY + kk4 * 4) = src64[(long)r * (C_IN / 4) + kk4];
        }
    }
    __syncthreads();

    const v8f zero8 = {0.f, 0.f, 0.f, 0.f, 0.f, 0.f, 0.f, 0.f};
    v8f acc2[12];
    #pragma unroll
    for (int t = 0; t < 12; ++t) acc2[t] = zero8;

    const int arow  = mw * 16 + (lane & 15);  // A-fragment row
    const int akoff = (lane >> 4) * 16;       // A-fragment K sub-offset
    const int bcol  = lane & 15;              // B-fragment column lane
    const int bkoff = (lane >> 4) * 16;       // B-fragment K sub-offset
    const int mbase = mw * 16 + (lane >> 4) * 8;  // C/D row base for this lane

    for (int j = 0; j < C_HID / NH; ++j) {
        // ---- matmul1: h_chunk = y_tile(64x384) @ w1t-cols(384x64) ----
        v8f acc1[2] = {zero8, zero8};
        for (int kb = 0; kb < C_IN; kb += 32) {
            const __bf16* ap = y_tile + arow * LDY + kb + akoff;
            v8bf alo = *(const v8bf*)(ap);
            v8bf ahi = *(const v8bf*)(ap + 8);
            v16bf a;
            #pragma unroll
            for (int e = 0; e < 8; ++e) { a[e] = alo[e]; a[8 + e] = ahi[e]; }
            #pragma unroll
            for (int t = 0; t < 2; ++t) {
                int n = j * NH + nw * 32 + t * 16 + bcol;
                v16bf bf = *(const v16bf*)(w1t + (long)n * C_IN + kb + bkoff);
                acc1[t] = __builtin_amdgcn_wmma_f32_16x16x32_bf16(
                    false, a, false, bf, (short)0, acc1[t], false, false);
            }
        }
        // ---- bias + GELU -> bf16 h_tile ----
        #pragma unroll
        for (int t = 0; t < 2; ++t) {
            int nloc = nw * 32 + t * 16 + bcol;
            float bias = b1[j * NH + nloc];
            #pragma unroll
            for (int v = 0; v < 8; ++v) {
                float g = gelu_tanh(acc1[t][v] + bias);
                h_tile[(mbase + v) * LDH + nloc] = (__bf16)g;
            }
        }
        __syncthreads();
        // ---- matmul2 accumulate: out64x384 += h_tile(64x64) @ w2t-rows(64x384) ----
        #pragma unroll
        for (int kb = 0; kb < NH; kb += 32) {
            const __bf16* ap = h_tile + arow * LDH + kb + akoff;
            v8bf alo = *(const v8bf*)(ap);
            v8bf ahi = *(const v8bf*)(ap + 8);
            v16bf a;
            #pragma unroll
            for (int e = 0; e < 8; ++e) { a[e] = alo[e]; a[8 + e] = ahi[e]; }
            #pragma unroll
            for (int t = 0; t < 12; ++t) {
                int n = nw * 192 + t * 16 + bcol;
                v16bf bf = *(const v16bf*)(w2t + (long)n * C_HID + j * NH + kb + bkoff);
                acc2[t] = __builtin_amdgcn_wmma_f32_16x16x32_bf16(
                    false, a, false, bf, (short)0, acc2[t], false, false);
            }
        }
        __syncthreads();
    }

    // ---- epilogue: +b2, *gamma, +x residual ----
    #pragma unroll
    for (int t = 0; t < 12; ++t) {
        int n = nw * 192 + t * 16 + bcol;
        float bb = b2[n];
        float gg = gamma[n];
        #pragma unroll
        for (int v = 0; v < 8; ++v) {
            long idx = (tok0 + mbase + v) * C_IN + n;
            out[idx] = x[idx] + (acc2[t][v] + bb) * gg;
        }
    }
}

// ---------------- launcher ----------------
extern "C" void kernel_launch(void* const* d_in, const int* in_sizes, int n_in,
                              void* d_out, int out_size, void* d_ws, size_t ws_size,
                              hipStream_t stream) {
    const float* x     = (const float*)d_in[0];
    const float* dwk   = (const float*)d_in[1];
    const float* dwb   = (const float*)d_in[2];
    const float* lns   = (const float*)d_in[3];
    const float* lnb   = (const float*)d_in[4];
    const float* w1    = (const float*)d_in[5];
    const float* b1    = (const float*)d_in[6];
    const float* w2    = (const float*)d_in[7];
    const float* b2    = (const float*)d_in[8];
    const float* gamma = (const float*)d_in[9];
    float* out = (float*)d_out;

    char* ws = (char*)d_ws;
    size_t y_bytes = (size_t)NTOK * C_IN * sizeof(__bf16);         // ~36.8 MB
    size_t w_bytes = (size_t)C_IN * C_HID * sizeof(__bf16);        // 1.18 MB each
    __bf16* y_bf  = (__bf16*)ws;
    __bf16* w1t   = (__bf16*)(ws + y_bytes);
    __bf16* w2t   = (__bf16*)(ws + y_bytes + w_bytes);

    int wele = C_IN * C_HID;
    transpose_w1_kernel<<<(wele + 255) / 256, 256, 0, stream>>>(w1, w1t);
    transpose_w2_kernel<<<(wele + 255) / 256, 256, 0, stream>>>(w2, w2t);
    dwconv_ln_kernel<<<NTOK, 384, 0, stream>>>(x, dwk, dwb, lns, lnb, y_bf);
    mlp_wmma_kernel<<<NTOK / MTILE, 256, 0, stream>>>(y_bf, w1t, b1, w2t, b2, gamma, x, out);
}
